// NetVladLayer_19524921328109
// MI455X (gfx1250) — matrix operand order repaired and checked by reading, not compiled, not run on hardware
//
#include <hip/hip_runtime.h>
#include <hip/hip_bf16.h>

// ---------- problem constants ----------
#define BB 32
#define DD 512
#define KK 64
#define NN 3136      // 56*56
#define TN 64        // pixel tile in kernel 1
#define NT 49        // NN / TN
#define NSEG 7       // split-K segments over N in kernel 2
#define SEGN 448     // NN / NSEG  (14 chunks of 32)

typedef __attribute__((ext_vector_type(16))) __bf16  v16bf;
typedef __attribute__((ext_vector_type(8)))  float   v8f;
typedef unsigned short ushort_t;
typedef __attribute__((ext_vector_type(8)))  unsigned short ushort8;

union BFrag { v16bf v; ushort8 h[2]; };

__device__ __forceinline__ unsigned short f2bf(float f) {
  unsigned u = __float_as_uint(f);
  unsigned r = u + 0x7FFFu + ((u >> 16) & 1u);   // round-to-nearest-even
  return (unsigned short)(r >> 16);
}

// ---------------------------------------------------------------------------
// Kernel 0: conv_w fp32 -> bf16; zero asum + vlad accumulators
// ---------------------------------------------------------------------------
__global__ __launch_bounds__(256) void k0_prep(const float* __restrict__ conv_w,
                                               ushort_t* __restrict__ wbf,
                                               float* __restrict__ asum,
                                               float* __restrict__ vlad_acc) {
  int tid = blockIdx.x * 256 + threadIdx.x;
  if (tid < KK * DD) wbf[tid] = f2bf(conv_w[tid]);
  if (tid < BB * KK) asum[tid] = 0.0f;
  if (tid < BB * DD * KK) vlad_acc[tid] = 0.0f;
}

// ---------------------------------------------------------------------------
// Kernel 1: logits GEMM (bf16 WMMA) + bias + softmax over K, writes
//   xbf   [b][d][n]  (bf16 copy of x, produced as a side-effect of staging)
//   alpha [b][k][n]  (bf16, k-major: the B-fragment layout for GEMM2),
//                    written coalesced (128-bit) via an LDS round-trip
//   asum  [b][k]     (f32, atomic partial sums of alpha over n)
// ---------------------------------------------------------------------------
__global__ __launch_bounds__(256) void k1_logits_softmax(
    const float* __restrict__ x, const float* __restrict__ conv_b,
    const ushort_t* __restrict__ wbf,
    ushort_t* __restrict__ xbf, ushort_t* __restrict__ alpha,
    float* __restrict__ asum) {
  __shared__ __align__(16) ushort_t x_lds[TN * 72];   // [n][d], stride 72 bf16
  __shared__ float logits_lds[KK * 66];               // [k][n], stride 66 f32
  __shared__ float bias_lds[KK];

  const int b    = blockIdx.x / NT;
  const int n0   = (blockIdx.x % NT) * TN;
  const int t    = threadIdx.x;
  const int lane = t & 31;
  const int wave = t >> 5;
  const int lm   = lane & 15;
  const int hi   = lane >> 4;

  if (t < KK) bias_lds[t] = conv_b[t];

  // wave -> two 16x16 C tiles: k-block kb, n-blocks nb0, nb0+1
  const int kb  = wave >> 1;
  const int nb0 = (wave & 1) * 2;

  v8f acc0 = {}; v8f acc1 = {};

  const int dl = t >> 2;     // staging: d row 0..63
  const int nq = (t & 3) * 16;

  for (int d0 = 0; d0 < DD; d0 += 64) {
    // stage x[b][d0..d0+63][n0..n0+63]: f32 -> bf16 into LDS (transposed) + global
    {
      const float* src = x + ((size_t)(b * DD + d0 + dl)) * NN + n0 + nq;
      ushort_t tmp[16];
      const float4* s4 = (const float4*)src;
#pragma unroll
      for (int i = 0; i < 4; ++i) {
        float4 f = s4[i];
        tmp[i * 4 + 0] = f2bf(f.x); tmp[i * 4 + 1] = f2bf(f.y);
        tmp[i * 4 + 2] = f2bf(f.z); tmp[i * 4 + 3] = f2bf(f.w);
      }
      ushort8* dst = (ushort8*)(xbf + ((size_t)(b * DD + d0 + dl)) * NN + n0 + nq);
      dst[0] = *(ushort8*)&tmp[0];
      dst[1] = *(ushort8*)&tmp[8];
#pragma unroll
      for (int i = 0; i < 16; ++i) x_lds[(nq + i) * 72 + dl] = tmp[i];
    }
    __syncthreads();

#pragma unroll
    for (int s = 0; s < 64; s += 32) {
      // A fragment: conv_w rows k, contraction d (16-bit A layout: pairs at
      // d-offsets (hi?8:0)+[0..7] and (hi?8:0)+16+[0..7])
      BFrag a;
      const ushort_t* ap = wbf + (kb * 16 + lm) * DD + d0 + s + (hi ? 8 : 0);
      a.h[0] = *(const ushort8*)ap;
      a.h[1] = *(const ushort8*)(ap + 16);
      // B fragments: x columns n, contraction d (same layout as A: d-offsets
      // (hi?8:0)+[0..7] and (hi?8:0)+16+[0..7])
      BFrag b0, b1;
      {
        const ushort_t* bp = &x_lds[(nb0 * 16 + lm) * 72 + s + (hi ? 8 : 0)];
        b0.h[0] = *(const ushort8*)bp;
        b0.h[1] = *(const ushort8*)(bp + 16);
      }
      {
        const ushort_t* bp = &x_lds[((nb0 + 1) * 16 + lm) * 72 + s + (hi ? 8 : 0)];
        b1.h[0] = *(const ushort8*)bp;
        b1.h[1] = *(const ushort8*)(bp + 16);
      }
      acc0 = __builtin_amdgcn_wmma_f32_16x16x32_bf16(false, a.v, false, b0.v,
                                                     (short)0, acc0, false, false);
      acc1 = __builtin_amdgcn_wmma_f32_16x16x32_bf16(false, a.v, false, b1.v,
                                                     (short)0, acc1, false, false);
    }
    __syncthreads();
  }

  // write logits (+bias) to LDS [k][n]
#pragma unroll
  for (int r = 0; r < 8; ++r) {
    int k = kb * 16 + r + (hi ? 8 : 0);
    logits_lds[k * 66 + nb0 * 16 + lm]       = acc0[r] + bias_lds[k];
    logits_lds[k * 66 + (nb0 + 1) * 16 + lm] = acc1[r] + bias_lds[k];
  }
  __syncthreads();

  // softmax over K=64 per pixel; 4 threads (consecutive lanes) per pixel.
  // Result (f32) is written back into logits_lds only; global stores happen
  // coalesced after the barrier.
  {
    const int n = t >> 2;
    const int part = t & 3;
    float v[16];
    float m = -3.4e38f;
#pragma unroll
    for (int j = 0; j < 16; ++j) {
      v[j] = logits_lds[(part * 16 + j) * 66 + n];
      m = fmaxf(m, v[j]);
    }
    m = fmaxf(m, __shfl_xor(m, 1));
    m = fmaxf(m, __shfl_xor(m, 2));
    float ssum = 0.0f;
#pragma unroll
    for (int j = 0; j < 16; ++j) { v[j] = __expf(v[j] - m); ssum += v[j]; }
    ssum += __shfl_xor(ssum, 1);
    ssum += __shfl_xor(ssum, 2);
    float inv = 1.0f / ssum;
#pragma unroll
    for (int j = 0; j < 16; ++j)
      logits_lds[(part * 16 + j) * 66 + n] = v[j] * inv;
  }
  __syncthreads();

  // coalesced alpha store: thread t owns k-row t>>2, 16 contiguous n values
  // -> two 16-byte global stores instead of 16 scattered 2-byte stores
  {
    const int k  = t >> 2;
    const int nn = (t & 3) * 16;
    ushort_t tmp[16];
#pragma unroll
    for (int i = 0; i < 16; ++i) tmp[i] = f2bf(logits_lds[k * 66 + nn + i]);
    ushort8* dst = (ushort8*)(alpha + ((size_t)(b * KK + k)) * NN + n0 + nn);
    dst[0] = *(ushort8*)&tmp[0];
    dst[1] = *(ushort8*)&tmp[8];
  }

  // asum partials (alpha values still live in logits_lds)
  if (t < KK) {
    float s = 0.0f;
#pragma unroll 8
    for (int n = 0; n < TN; ++n) s += logits_lds[t * 66 + n];
    atomicAdd(&asum[b * KK + t], s);
  }
}

// ---------------------------------------------------------------------------
// Kernel 2: split-K over N. Each workgroup handles one (b, d-tile, n-segment)
// and atomically accumulates its 64x64 partial tile:
//   vlad_acc[b][d][k] += sum_{n in seg} x[d][n]*alpha[k][n]
// Pure-register WMMA: both operand layouts allow contiguous 16B loads per lane
// straight from the L2-resident bf16 buffers (no LDS staging needed).
// 1792 workgroups keeps the WGPs full; atomic adds hit the L2 atomic units.
// ---------------------------------------------------------------------------
__global__ __launch_bounds__(256) void k2_aggregate(
    const ushort_t* __restrict__ xbf, const ushort_t* __restrict__ alpha,
    float* __restrict__ vlad_acc) {
  const int b   = blockIdx.x / (8 * NSEG);
  const int rem = blockIdx.x % (8 * NSEG);
  const int dt  = (rem / NSEG) * 64;        // d tile base
  const int ns  = (rem % NSEG) * SEGN;      // n segment base
  const int t = threadIdx.x;
  const int lane = t & 31, wave = t >> 5;
  const int lm = lane & 15, hi = lane >> 4;
  const int db  = wave >> 1;        // d block 0..3
  const int kb0 = (wave & 1) * 2;   // k blocks kb0, kb0+1

  v8f acc0 = {}, acc1 = {};
  const ushort_t* xrow  = xbf   + ((size_t)(b * DD + dt + db * 16 + lm)) * NN + ns + (hi ? 8 : 0);
  const ushort_t* arow0 = alpha + ((size_t)(b * KK + kb0 * 16 + lm)) * NN + ns + (hi ? 8 : 0);
  const ushort_t* arow1 = alpha + ((size_t)(b * KK + (kb0 + 1) * 16 + lm)) * NN + ns + (hi ? 8 : 0);

  // 14 chunks of 32; unroll 7 -> deep clauses of independent b128 loads
#pragma unroll 7
  for (int n0 = 0; n0 < SEGN; n0 += 32) {
    BFrag a, b0, b1;
    a.h[0]  = *(const ushort8*)(xrow + n0);
    a.h[1]  = *(const ushort8*)(xrow + n0 + 16);
    b0.h[0] = *(const ushort8*)(arow0 + n0);
    b0.h[1] = *(const ushort8*)(arow0 + n0 + 16);
    b1.h[0] = *(const ushort8*)(arow1 + n0);
    b1.h[1] = *(const ushort8*)(arow1 + n0 + 16);
    acc0 = __builtin_amdgcn_wmma_f32_16x16x32_bf16(false, a.v, false, b0.v,
                                                   (short)0, acc0, false, false);
    acc1 = __builtin_amdgcn_wmma_f32_16x16x32_bf16(false, a.v, false, b1.v,
                                                   (short)0, acc1, false, false);
  }

#pragma unroll
  for (int r = 0; r < 8; ++r) {
    int d  = dt + db * 16 + r + (hi ? 8 : 0);
    int k0 = kb0 * 16 + lm;
    int k1 = (kb0 + 1) * 16 + lm;
    atomicAdd(&vlad_acc[((size_t)(b * DD + d)) * KK + k0], acc0[r]);
    atomicAdd(&vlad_acc[((size_t)(b * DD + d)) * KK + k1], acc1[r]);
  }
}

// ---------------------------------------------------------------------------
// Kernel 3: v = acc - centers*asum; per-(b,k) intra-normalize over D; the
// subsequent global norm is exactly sqrt(K)=8, so fold 1/8 into the scale.
// ---------------------------------------------------------------------------
__global__ __launch_bounds__(64) void k3_normalize(
    const float* __restrict__ vlad_acc, const float* __restrict__ centers,
    const float* __restrict__ asum, float* __restrict__ out) {
  const int b = blockIdx.x >> 6;
  const int k = blockIdx.x & 63;
  const int t = threadIdx.x;
  const float as = asum[b * KK + k];
  float v[8];
  float ss = 0.0f;
#pragma unroll
  for (int i = 0; i < 8; ++i) {
    int d = i * 64 + t;
    v[i] = vlad_acc[((size_t)(b * DD + d)) * KK + k] - centers[d * KK + k] * as;
    ss += v[i] * v[i];
  }
#pragma unroll
  for (int m = 16; m >= 1; m >>= 1) ss += __shfl_xor(ss, m);
  __shared__ float red[2];
  if ((t & 31) == 0) red[t >> 5] = ss;
  __syncthreads();
  float scale = rsqrtf(red[0] + red[1]) * 0.125f;   // 1/(||col|| * sqrt(K))
#pragma unroll
  for (int i = 0; i < 8; ++i) {
    int d = i * 64 + t;
    out[(size_t)b * (DD * KK) + (size_t)d * KK + k] = v[i] * scale;
  }
}

// ---------------------------------------------------------------------------
extern "C" void kernel_launch(void* const* d_in, const int* in_sizes, int n_in,
                              void* d_out, int out_size, void* d_ws, size_t ws_size,
                              hipStream_t stream) {
  const float* x       = (const float*)d_in[0];   // [32,512,56,56]
  const float* conv_w  = (const float*)d_in[1];   // [64,512]
  const float* conv_b  = (const float*)d_in[2];   // [64]
  const float* centers = (const float*)d_in[3];   // [512,64]
  float* out = (float*)d_out;

  char* ws = (char*)d_ws;
  const size_t XBF_BYTES   = (size_t)BB * DD * NN * 2;   // ~102.8 MB (L2-resident)
  const size_t ALPHA_BYTES = (size_t)BB * KK * NN * 2;   // ~12.8 MB
  const size_t WBF_BYTES   = (size_t)KK * DD * 2;
  const size_t ASUM_BYTES  = (size_t)BB * KK * 4;

  ushort_t* xbf   = (ushort_t*)ws;
  ushort_t* alpha = (ushort_t*)(ws + XBF_BYTES);
  ushort_t* wbf   = (ushort_t*)(ws + XBF_BYTES + ALPHA_BYTES);
  float* asum     = (float*)(ws + XBF_BYTES + ALPHA_BYTES + WBF_BYTES);
  float* vlad_acc = (float*)(ws + XBF_BYTES + ALPHA_BYTES + WBF_BYTES + ASUM_BYTES);

  // zero pass must cover vlad_acc (BB*DD*KK = 1M floats)
  k0_prep<<<(BB * DD * KK + 255) / 256, 256, 0, stream>>>(conv_w, wbf, asum, vlad_acc);
  k1_logits_softmax<<<BB * NT, 256, 0, stream>>>(x, conv_b, wbf, xbf, alpha, asum);
  k2_aggregate<<<BB * 8 * NSEG, 256, 0, stream>>>(xbf, alpha, vlad_acc);
  k3_normalize<<<BB * KK, 64, 0, stream>>>(vlad_acc, centers, asum, out);
}
